// BitLinear_16647293239422
// MI455X (gfx1250) — matrix operand order repaired and checked by reading, not compiled, not run on hardware
//
#include <hip/hip_runtime.h>
#include <math.h>

typedef __bf16 bf16_t;
typedef __attribute__((ext_vector_type(16))) __bf16 v16bf;
typedef __attribute__((ext_vector_type(8)))  __bf16 v8bf;
typedef __attribute__((ext_vector_type(4)))  __bf16 v4bf;
typedef __attribute__((ext_vector_type(8)))  float  v8f;
typedef __attribute__((ext_vector_type(4)))  float  v4f;

#define LN_THREADS 256

// ---------------------------------------------------------------------------
// CDNA5 async global->LDS path (ASYNCcnt-tracked), with sync fallback.
// Builtin pointee type is vector_size(16) int (from hipcc diagnostic).
// ---------------------------------------------------------------------------
#if __has_builtin(__builtin_amdgcn_global_load_async_to_lds_b128)
#define HAVE_ASYNC_LDS 1
#else
#define HAVE_ASYNC_LDS 0
#endif

typedef int v4i_vs __attribute__((vector_size(4 * sizeof(int))));
typedef __attribute__((address_space(1))) v4i_vs* g_v4i_ptr;
typedef __attribute__((address_space(3))) v4i_vs* l_v4i_ptr;

__device__ __forceinline__ void copy16_g2lds(bf16_t* dst_lds, const bf16_t* src_g)
{
#if HAVE_ASYNC_LDS
    __builtin_amdgcn_global_load_async_to_lds_b128(
        (g_v4i_ptr)src_g, (l_v4i_ptr)dst_lds, /*offset=*/0, /*cpol=*/0);
#else
    *(v8bf*)dst_lds = *(const v8bf*)src_g;   // load + ds_store fallback
#endif
}

template <int N>
__device__ __forceinline__ void wait_async()
{
#if HAVE_ASYNC_LDS
#if __has_builtin(__builtin_amdgcn_s_wait_asynccnt)
    __builtin_amdgcn_s_wait_asynccnt(N);
#else
    asm volatile("s_wait_asynccnt %0" :: "n"(N) : "memory");
#endif
#endif
}

__device__ __forceinline__ void wait_ds0()
{
#if __has_builtin(__builtin_amdgcn_s_wait_dscnt)
    __builtin_amdgcn_s_wait_dscnt(0);
#else
    asm volatile("s_wait_dscnt 0x0" ::: "memory");
#endif
}

// ---------------------------------------------------------------------------
// Kernel 1: row-wise LayerNorm (torch semantics: unbiased std, /(std+eps)),
// output bf16. One 256-thread block per token row (K = 4096).
// ---------------------------------------------------------------------------
__global__ void __launch_bounds__(LN_THREADS)
ln_to_bf16_kernel(const float* __restrict__ x, bf16_t* __restrict__ xn, int K)
{
    __shared__ float red_s[LN_THREADS];
    __shared__ float red_q[LN_THREADS];
    const int tid = threadIdx.x;
    const size_t row_off = (size_t)blockIdx.x * (size_t)K;
    const float* row = x + row_off;

    v4f vals[4];
    float s = 0.f, q = 0.f;
#pragma unroll
    for (int j = 0; j < 4; ++j) {
        vals[j] = *(const v4f*)(row + tid * 4 + j * (LN_THREADS * 4));
#pragma unroll
        for (int c = 0; c < 4; ++c) { float v = vals[j][c]; s += v; q += v * v; }
    }
    red_s[tid] = s;
    red_q[tid] = q;
    __syncthreads();
#pragma unroll
    for (int off = LN_THREADS / 2; off > 0; off >>= 1) {
        if (tid < off) {
            red_s[tid] += red_s[tid + off];
            red_q[tid] += red_q[tid + off];
        }
        __syncthreads();
    }
    const float sum   = red_s[0];
    const float sumsq = red_q[0];
    const float mean  = sum / (float)K;
    float var = (sumsq - sum * mean) / (float)(K - 1);   // unbiased (ddof=1)
    var = fmaxf(var, 0.f);
    const float inv = 1.f / (sqrtf(var) + 1e-5f);        // /(std + eps)

    bf16_t* orow = xn + row_off;
#pragma unroll
    for (int j = 0; j < 4; ++j) {
        v4bf o;
#pragma unroll
        for (int c = 0; c < 4; ++c) o[c] = (bf16_t)((vals[j][c] - mean) * inv);
        *(v4bf*)(orow + tid * 4 + j * (LN_THREADS * 4)) = o;
    }
}

// ---------------------------------------------------------------------------
// Kernel 2: sign(weight) -> bf16 {-1, 0, +1} (exact in bf16)
// ---------------------------------------------------------------------------
__global__ void __launch_bounds__(256)
sign_to_bf16_kernel(const float* __restrict__ w, bf16_t* __restrict__ ws, size_t n)
{
    const size_t i = ((size_t)blockIdx.x * 256 + threadIdx.x) * 4;
    if (i + 3 < n) {
        v4f v = *(const v4f*)(w + i);
        v4bf o;
#pragma unroll
        for (int c = 0; c < 4; ++c) {
            float sv = (v[c] > 0.f) ? 1.f : ((v[c] < 0.f) ? -1.f : 0.f);
            o[c] = (bf16_t)sv;
        }
        *(v4bf*)(ws + i) = o;
    }
}

// ---------------------------------------------------------------------------
// Kernel 3: out = xn @ sign(W)^T + bias via v_wmma_f32_16x16x32_bf16, with
// double-buffered async global->LDS staging (global_load_async_to_lds_b128 /
// s_wait_asynccnt) and LDS fragment reads (ds_load_b128 / s_wait_dscnt).
//
// 256 threads = 8 waves, arranged 2(M) x 4(N). Block tile 64(M) x 256(N).
// Wave tile 32x64 = 2x4 accumulators -> 8 WMMA per K-step of 32.
// Per K-step the block stages A 64x32 (4 KB) + B 256x32 (16 KB) into LDS:
// 20 KB of global traffic for 64 WMMAs (~1 MFLOP) => ~52 FLOP/B off HBM/L2.
// Last K-step is peeled so the steady-state loop is branch-free.
//
// LDS rows padded to 40 bf16 (80 B = 5 banks*16B) to spread ds_load_b128
// accesses across banks. Double buffer: 2*(64+256)*40*2B = 50 KB.
//
// Fragment layouts (ISA 7.12.2, wave32):
//   A 16x32 bf16: lane l -> m=l%16, hl=l/16; K runs [hl*8,+8) and [16+hl*8,+8)
//   B 32x16 bf16: lane l -> n=l%16, hl=l/16; K runs [hl*8,+8) and [16+hl*8,+8)
//   C 16x16 f32:  VGPR r -> m = r + 8*hl, n = l%16
// ---------------------------------------------------------------------------
#define APAD 40
#define BPAD 40

__global__ void __launch_bounds__(256)
bitlinear_wmma_gemm(const bf16_t* __restrict__ A,    // [T,K] bf16 (layernormed)
                    const bf16_t* __restrict__ B,    // [O,K] bf16 (signs)
                    const float*  __restrict__ bias, // [O]
                    float* __restrict__ out,         // [T,O]
                    int T, int K, int O)
{
    __shared__ bf16_t lA[2][64 * APAD];
    __shared__ bf16_t lB[2][256 * BPAD];

    const int tid   = threadIdx.x;
    const int lane  = tid & 31;
    const int wave  = tid >> 5;
    const int wm    = wave & 1;       // 2 waves along M
    const int wn    = wave >> 1;      // 4 waves along N
    const int mBlk  = blockIdx.x * 64;
    const int nBlk  = blockIdx.y * 256;
    const int mlane = lane & 15;
    const int hl    = lane >> 4;

    // async-copy chunk mapping (16 B per lane per instruction)
    const int aRow = tid >> 2, aCol = (tid & 3) * 8;

    auto load_tile = [&](int buf, int k0) {
        // A tile: 64 rows x 32 k = 256 chunks, one per thread
        copy16_g2lds(&lA[buf][aRow * APAD + aCol],
                     A + (size_t)(mBlk + aRow) * K + k0 + aCol);
        // B tile: 256 rows x 32 k = 1024 chunks, four per thread
#pragma unroll
        for (int i = 0; i < 4; ++i) {
            const int chunk = i * 256 + tid;
            const int row = chunk >> 2, col = (chunk & 3) * 8;
            copy16_g2lds(&lB[buf][row * BPAD + col],
                         B + (size_t)(nBlk + row) * K + k0 + col);
        }
    };

    v8f acc[2][4] = {};

    auto compute_step = [&](int buf) {
        v16bf afrag[2];
#pragma unroll
        for (int mt = 0; mt < 2; ++mt) {
            const bf16_t* ar = &lA[buf][(wm * 32 + mt * 16 + mlane) * APAD];
            v8bf lo = *(const v8bf*)(ar + hl * 8);
            v8bf hi = *(const v8bf*)(ar + 16 + hl * 8);
            afrag[mt] = __builtin_shufflevector(lo, hi,
                                                0, 1, 2, 3, 4, 5, 6, 7,
                                                8, 9, 10, 11, 12, 13, 14, 15);
        }
#pragma unroll
        for (int nt = 0; nt < 4; ++nt) {
            const bf16_t* br =
                &lB[buf][(wn * 64 + nt * 16 + mlane) * BPAD];
            v8bf blo = *(const v8bf*)(br + hl * 8);
            v8bf bhi = *(const v8bf*)(br + 16 + hl * 8);
            v16bf bfrag = __builtin_shufflevector(blo, bhi,
                                                  0, 1, 2, 3, 4, 5, 6, 7,
                                                  8, 9, 10, 11, 12, 13, 14, 15);
            acc[0][nt] = __builtin_amdgcn_wmma_f32_16x16x32_bf16(
                false, afrag[0], false, bfrag, (short)0, acc[0][nt], false, false);
            acc[1][nt] = __builtin_amdgcn_wmma_f32_16x16x32_bf16(
                false, afrag[1], false, bfrag, (short)0, acc[1][nt], false, false);
        }
    };

    const int ksteps = K / 32;
    load_tile(0, 0);                       // 5 async ops per wave in flight

    // steady state: branch-free, always prefetching the next buffer
    for (int kt = 0; kt < ksteps - 1; ++kt) {
        const int buf = kt & 1;
        load_tile(buf ^ 1, (kt + 1) * 32);
        wait_async<5>();                   // our 5 older ops (this buf) done
        __syncthreads();                   // everyone's chunks visible
        compute_step(buf);
        wait_ds0();                        // our LDS reads landed
        __syncthreads();                   // safe to overwrite this buffer
    }
    // peeled last step
    wait_async<0>();
    __syncthreads();
    compute_step((ksteps - 1) & 1);

    // bias + store (non-temporal: out streams 537 MB once; keep bf16 operands
    // resident in the 192 MB L2)
    float bb[4];
#pragma unroll
    for (int nt = 0; nt < 4; ++nt)
        bb[nt] = bias[nBlk + wn * 64 + nt * 16 + mlane];

#pragma unroll
    for (int mt = 0; mt < 2; ++mt) {
        const size_t obase =
            (size_t)(mBlk + wm * 32 + mt * 16 + 8 * hl) * (size_t)O
            + (size_t)nBlk + wn * 64 + mlane;
#pragma unroll
        for (int r = 0; r < 8; ++r) {
            float* po = out + obase + (size_t)r * (size_t)O;
#pragma unroll
            for (int nt = 0; nt < 4; ++nt)
                __builtin_nontemporal_store(acc[mt][nt][r] + bb[nt], po + nt * 16);
        }
    }
}

// ---------------------------------------------------------------------------
extern "C" void kernel_launch(void* const* d_in, const int* in_sizes, int n_in,
                              void* d_out, int out_size, void* d_ws, size_t ws_size,
                              hipStream_t stream)
{
    const float* x    = (const float*)d_in[0];
    const float* w    = (const float*)d_in[1];
    const float* bias = (const float*)d_in[2];
    float* out = (float*)d_out;

    const int O = in_sizes[2];          // 16384
    const int K = in_sizes[1] / O;      // 4096
    const int T = in_sizes[0] / K;      // 8192

    // workspace: bf16 xn [T,K] then bf16 sign(W) [O,K]  (~201 MB total)
    bf16_t* xn   = (bf16_t*)d_ws;
    bf16_t* wsgn = xn + (size_t)T * (size_t)K;

    ln_to_bf16_kernel<<<T, LN_THREADS, 0, stream>>>(x, xn, K);

    const size_t nW = (size_t)O * (size_t)K;
    sign_to_bf16_kernel<<<(unsigned)(nW / (256 * 4)), 256, 0, stream>>>(w, wsgn, nW);

    dim3 grid(T / 64, O / 256);
    bitlinear_wmma_gemm<<<grid, 256, 0, stream>>>(xn, wsgn, bias, out, T, K, O);
}
